// MultiHeadAttention_13185549599174
// MI455X (gfx1250) — matrix operand order repaired and checked by reading, not compiled, not run on hardware
//
#include <hip/hip_runtime.h>

// ---------------------------------------------------------------------------
// Sliding-window GQA attention (T=4096, DM=2048, HQ=32, HK=8, HD=64, W=512)
// f16 WMMA (v_wmma_f32_16x16x32_f16) everywhere; flash attention computed as
// S^T = K·Q^T so softmax stats live on the lane axis (wave32 C-layout) and the
// P fragment for O^T += V^T·P^T is built from own registers, no shuffle or LDS transpose.
// GEMM: one wave = 32x64 output tile -> 12 b128 loads : 8 wmma per k-step.
// ---------------------------------------------------------------------------

typedef _Float16 half8 __attribute__((ext_vector_type(8)));
typedef _Float16 v16h  __attribute__((ext_vector_type(16)));
typedef float    v8f   __attribute__((ext_vector_type(8)));

constexpr int T   = 4096;
constexpr int DM  = 2048;
constexpr int HQ  = 32;
constexpr int HK  = 8;
constexpr int HD  = 64;
constexpr int WIN = 512;
constexpr int TP  = T + 32;        // keys padded so 32-wide tiles never read OOB
constexpr int QLD = HQ * HD;       // 2048
constexpr int KLD = HK * HD;       // 512

#define DEVFN static __device__ __forceinline__

DEVFN v16h cat8(half8 lo, half8 hi) {
  return __builtin_shufflevector(lo, hi, 0,1,2,3,4,5,6,7,8,9,10,11,12,13,14,15);
}

// A-matrix fragment (16x32 f16, ISA 7.12.2): lane holds row m=lane&15;
// halves j=0..7 -> k = 8*(lane>=16)+j ; j=8..15 -> k = 16+8*(lane>=16)+(j-8).
// rowptr must already point at &A[row0 + (lane&15)][col0].
DEVFN v16h lda_frag(const _Float16* rowptr, int lane) {
  const int hh = (lane >> 4) * 8;
  half8 lo = *(const half8*)(rowptr + hh);
  half8 hi = *(const half8*)(rowptr + hh + 16);
  return cat8(lo, hi);
}

// B-matrix fragment (32x16 f16): lane holds col n=lane&15, same k order as lda_frag.
// rowptr points at &W[n0 + (lane&15)][k0] for W stored row-major over k.
DEVFN v16h ldb_frag(const _Float16* rowptr, int lane) {
  const int hh = (lane >> 4) * 8;
  half8 lo = *(const half8*)(rowptr + hh);
  half8 hi = *(const half8*)(rowptr + hh + 16);
  return cat8(lo, hi);
}

DEVFN v8f wmma16(v16h a, v16h b, v8f c) {
  return __builtin_amdgcn_wmma_f32_16x16x32_f16(false, a, false, b, (short)0, c,
                                                false, false);
}

// ---------------------------------------------------------------------------
// f32 -> f16 elementwise convert (4 elems / thread)
// ---------------------------------------------------------------------------
__global__ __launch_bounds__(256) void cvt_f32_f16(const float* __restrict__ src,
                                                   _Float16* __restrict__ dst,
                                                   int n) {
  int i = (blockIdx.x * blockDim.x + threadIdx.x) * 4;
  if (i + 3 < n) {
    float4 v = *(const float4*)(src + i);
    dst[i + 0] = (_Float16)v.x;
    dst[i + 1] = (_Float16)v.y;
    dst[i + 2] = (_Float16)v.z;
    dst[i + 3] = (_Float16)v.w;
  }
}

// ---------------------------------------------------------------------------
// WMMA GEMM: C[m][n] = sum_k A[m][k] * Bw[n][k]   (A: MxK f16, Bw: NxK f16)
// One wave = 32x64 output tile (2x4 accumulators), K-loop step 32.
// mode 0: f32 row-major (ldc)   mode 1: f16 row-major (ldc)
// mode 2: f16 transposed store  C^T[n][m] with column stride ldc
// ---------------------------------------------------------------------------
__global__ __launch_bounds__(256) void gemm_f16_wmma(
    const _Float16* __restrict__ A, const _Float16* __restrict__ Bw,
    void* __restrict__ Cout, int M, int N, int K, int ldc, int mode) {
  const int lane = threadIdx.x & 31;
  const int wid  = blockIdx.x * (blockDim.x >> 5) + (threadIdx.x >> 5);
  const int tiles_m = M >> 5;
  const int tm = wid % tiles_m;
  const int tn = wid / tiles_m;
  if (tn >= (N >> 6)) return;
  const int m0 = tm << 5, n0 = tn << 6;
  const int nl = lane & 15;

  const _Float16* arow0 = A + (size_t)(m0 + nl) * K;
  const _Float16* arow1 = arow0 + (size_t)16 * K;
  const _Float16* brow0 = Bw + (size_t)(n0 + nl) * K;
  const _Float16* brow1 = brow0 + (size_t)16 * K;
  const _Float16* brow2 = brow0 + (size_t)32 * K;
  const _Float16* brow3 = brow0 + (size_t)48 * K;

  v8f c00 = {}, c01 = {}, c02 = {}, c03 = {};
  v8f c10 = {}, c11 = {}, c12 = {}, c13 = {};
  for (int k = 0; k < K; k += 32) {
    v16h a0 = lda_frag(arow0 + k, lane);
    v16h a1 = lda_frag(arow1 + k, lane);
    v16h b0 = ldb_frag(brow0 + k, lane);
    v16h b1 = ldb_frag(brow1 + k, lane);
    v16h b2 = ldb_frag(brow2 + k, lane);
    v16h b3 = ldb_frag(brow3 + k, lane);
    c00 = wmma16(a0, b0, c00);
    c01 = wmma16(a0, b1, c01);
    c02 = wmma16(a0, b2, c02);
    c03 = wmma16(a0, b3, c03);
    c10 = wmma16(a1, b0, c10);
    c11 = wmma16(a1, b1, c11);
    c12 = wmma16(a1, b2, c12);
    c13 = wmma16(a1, b3, c13);
  }

  // C layout: vgpr r, lane L -> m = r + 8*(L>=16), n = L&15
  const int hh8 = (lane >> 4) << 3;
  float*    cf = (float*)Cout;
  _Float16* ch = (_Float16*)Cout;
#pragma unroll
  for (int ti = 0; ti < 8; ++ti) {
    v8f acc;
    switch (ti) {
      case 0: acc = c00; break; case 1: acc = c01; break;
      case 2: acc = c02; break; case 3: acc = c03; break;
      case 4: acc = c10; break; case 5: acc = c11; break;
      case 6: acc = c12; break; default: acc = c13; break;
    }
    const int mi = (ti >> 2) * 16, ni = (ti & 3) * 16;
#pragma unroll
    for (int r = 0; r < 8; ++r) {
      const int row = m0 + mi + r + hh8;
      const int col = n0 + ni + nl;
      if (mode == 0)      cf[(size_t)row * ldc + col] = acc[r];
      else if (mode == 1) ch[(size_t)row * ldc + col] = (_Float16)acc[r];
      else                ch[(size_t)col * ldc + row] = (_Float16)acc[r];
    }
  }
}

// ---------------------------------------------------------------------------
// Fused per-head RMSNorm + half-split RoPE, in-place on f16 buffer.
// One wave = one (t, head); lane i owns the RoPE pair (i, i+32).
// Rows t >= tvalid (K padding) are zero-filled.
// ---------------------------------------------------------------------------
__global__ __launch_bounds__(128) void norm_rope_f16(
    _Float16* __restrict__ buf, const int* __restrict__ pos,
    const float* __restrict__ w, int nheads, int ld, int tvalid, int ttotal) {
  const int lane = threadIdx.x & 31;
  const int wid  = blockIdx.x * (blockDim.x >> 5) + (threadIdx.x >> 5);
  if (wid >= nheads * ttotal) return;
  const int t = wid / nheads;
  const int h = wid - t * nheads;
  _Float16* p = buf + (size_t)t * ld + h * HD;
  if (t >= tvalid) { p[lane] = (_Float16)0.f; p[lane + 32] = (_Float16)0.f; return; }

  float x1 = (float)p[lane];
  float x2 = (float)p[lane + 32];
  float ss = x1 * x1 + x2 * x2;
#pragma unroll
  for (int m = 16; m >= 1; m >>= 1) ss += __shfl_xor(ss, m, 32);
  const float rn = rsqrtf(ss * (1.0f / 64.0f) + 1e-6f);
  const float y1 = x1 * rn * w[lane];
  const float y2 = x2 * rn * w[lane + 32];
  // inv_freq = 10000^(-lane/32); ln(10000) = 9.210340372
  const float ang = (float)pos[t] * __expf(-(float)lane * (9.210340371976184f / 32.0f));
  const float c = cosf(ang), s = sinf(ang);
  p[lane]      = (_Float16)(y1 * c - y2 * s);
  p[lane + 32] = (_Float16)(y2 * c + y1 * s);
}

// Zero the 32 padded key columns of V^T.
__global__ __launch_bounds__(256) void vpad_zero(_Float16* __restrict__ vt) {
  const int i = blockIdx.x * blockDim.x + threadIdx.x;   // 512*32 elements
  const int c = i >> 5, t = T + (i & 31);
  if (c < KLD) vt[(size_t)c * TP + t] = (_Float16)0.f;
}

// ---------------------------------------------------------------------------
// Flash sliding-window attention. One wave = 16 queries x 1 q-head.
//   S^T(keys x q) = K_tile(16k x 32hd) . Q^T(32hd x 16q)   [2 wmma / 16 keys]
//   O^T(hd x q)  += V^T(16hd x 32k)   . P^T(32k x 16q)     [4 wmma / 32 keys]
// Softmax stats are per-lane scalars (q = lane&15), reduced with shfl_xor(16).
// ---------------------------------------------------------------------------
__global__ __launch_bounds__(256) void attn_wmma(
    const _Float16* __restrict__ Q,   // [T][2048] f16 (normed+roped)
    const _Float16* __restrict__ Kb,  // [TP][512] f16 (normed+roped, pad=0)
    const _Float16* __restrict__ Vt,  // [512][TP] f16 transposed  (pad=0)
    _Float16* __restrict__ Ctx) {     // [T][2048] f16
  const int lane = threadIdx.x & 31;
  const int wid  = blockIdx.x * (blockDim.x >> 5) + (threadIdx.x >> 5);
  const int qt   = wid & 255;          // T/16 = 256 query tiles
  const int head = wid >> 8;           // 0..31
  if (head >= HQ) return;
  const int kvh = head >> 2;           // GQA group of 4
  const int t0  = qt * 16;
  const int nl  = lane & 15;
  const int hh8 = (lane >> 4) << 3;
  const int tq  = t0 + nl;             // this lane's query position

  // Q^T B-fragments (hd 0..31 and 32..63), loaded once.
  const _Float16* qrow = Q + (size_t)(t0 + nl) * QLD + head * HD + ((lane >> 4) << 3);
  const v16h bq0 = cat8(*(const half8*)(qrow),      *(const half8*)(qrow + 16));
  const v16h bq1 = cat8(*(const half8*)(qrow + 32), *(const half8*)(qrow + 48));

  v8f o0 = {}, o1 = {}, o2 = {}, o3 = {};
  float mrun = -1.0e30f, lrun = 0.0f;

  int klo = t0 - WIN; if (klo < 0) klo = 0;
  klo &= ~31;
  for (int kt = klo; kt <= t0 + 15; kt += 32) {
    // --- scores: S^T tiles for keys [kt,kt+16) and [kt+16,kt+32) ---
    const _Float16* kr0 = Kb + (size_t)(kt + nl) * KLD + kvh * HD;
    const _Float16* kr1 = kr0 + (size_t)16 * KLD;
    v8f s0 = {}, s1 = {};
    s0 = wmma16(lda_frag(kr0, lane),      bq0, s0);
    s0 = wmma16(lda_frag(kr0 + 32, lane), bq1, s0);
    s1 = wmma16(lda_frag(kr1, lane),      bq0, s1);
    s1 = wmma16(lda_frag(kr1 + 32, lane), bq1, s1);

    // --- scale + sliding-window causal mask (0 <= tq-tk <= WIN) ---
#pragma unroll
    for (int r = 0; r < 8; ++r) {
      const int tk = kt + r + hh8;
      const int d0 = tq - tk, d1 = d0 - 16;
      s0[r] = (d0 >= 0 && d0 <= WIN) ? s0[r] * 0.125f : -3.0e38f;
      s1[r] = (d1 >= 0 && d1 <= WIN) ? s1[r] * 0.125f : -3.0e38f;
    }

    // --- online softmax (stats per q = per lane, pairs share via xor 16) ---
    float tmax = s0[0];
#pragma unroll
    for (int r = 1; r < 8; ++r) tmax = fmaxf(tmax, s0[r]);
#pragma unroll
    for (int r = 0; r < 8; ++r) tmax = fmaxf(tmax, s1[r]);
    tmax = fmaxf(tmax, __shfl_xor(tmax, 16, 32));
    const float mnew  = fmaxf(mrun, tmax);
    const float alpha = __expf(mrun - mnew);
    mrun = mnew;

    v8f p0, p1;
    float tsum = 0.0f;
#pragma unroll
    for (int r = 0; r < 8; ++r) {
      p0[r] = __expf(s0[r] - mnew);
      p1[r] = __expf(s1[r] - mnew);
      tsum += p0[r] + p1[r];
    }
    tsum += __shfl_xor(tsum, 16, 32);
    lrun = lrun * alpha + tsum;
#pragma unroll
    for (int e = 0; e < 8; ++e) { o0[e] *= alpha; o1[e] *= alpha; o2[e] *= alpha; o3[e] *= alpha; }

    // --- P^T as B-fragment (32 keys x 16 q) from the two C-layout tiles ---
    // table k order: elems 0-7 = own tile0 rows, elems 8-15 = own tile1 rows.
    v16h pb;
    const bool hi = lane >= 16;
#pragma unroll
    for (int r = 0; r < 8; ++r) {
      const float sx0 = __shfl_xor(p0[r], 16, 32);
      const float sx1 = __shfl_xor(p1[r], 16, 32);
      pb[r]     = (_Float16)p0[r];
      pb[r + 8] = (_Float16)p1[r];
    }

    // --- O^T += V^T . P^T over the 4 hd tiles ---
    const _Float16* vr = Vt + (size_t)(kvh * HD + nl) * TP + kt;
    o0 = wmma16(lda_frag(vr, lane),                     pb, o0);
    o1 = wmma16(lda_frag(vr + (size_t)16 * TP, lane),   pb, o1);
    o2 = wmma16(lda_frag(vr + (size_t)32 * TP, lane),   pb, o2);
    o3 = wmma16(lda_frag(vr + (size_t)48 * TP, lane),   pb, o3);
  }

  // --- normalize by l and write ctx[t][head*64 + hd] (O^T: m=hd, n=q) ---
  const float inv = 1.0f / lrun;
  _Float16* crow = Ctx + (size_t)(t0 + nl) * QLD + head * HD;
#pragma unroll
  for (int d = 0; d < 4; ++d) {
    const v8f o = (d == 0) ? o0 : (d == 1) ? o1 : (d == 2) ? o2 : o3;
#pragma unroll
    for (int r = 0; r < 8; ++r) crow[d * 16 + r + hh8] = (_Float16)(o[r] * inv);
  }
}

// ---------------------------------------------------------------------------
// Host-side orchestration
// ---------------------------------------------------------------------------
extern "C" void kernel_launch(void* const* d_in, const int* in_sizes, int n_in,
                              void* d_out, int out_size, void* d_ws, size_t ws_size,
                              hipStream_t stream) {
  const float* x   = (const float*)d_in[0];
  const int*   pos = (const int*)d_in[1];
  const float* wq  = (const float*)d_in[2];
  const float* wk  = (const float*)d_in[3];
  const float* wv  = (const float*)d_in[4];
  const float* wo  = (const float*)d_in[5];
  const float* qnw = (const float*)d_in[6];
  const float* knw = (const float*)d_in[7];

  char* ws = (char*)d_ws;
  size_t off = 0;
  auto alloc = [&](size_t bytes) -> void* {
    void* p = ws + off;
    off = (off + bytes + 255) & ~(size_t)255;
    return p;
  };
  _Float16* xh   = (_Float16*)alloc((size_t)T * DM * 2);
  _Float16* wqh  = (_Float16*)alloc((size_t)QLD * DM * 2);
  _Float16* wkh  = (_Float16*)alloc((size_t)KLD * DM * 2);
  _Float16* wvh  = (_Float16*)alloc((size_t)KLD * DM * 2);
  _Float16* woh  = (_Float16*)alloc((size_t)DM * QLD * 2);
  _Float16* qh   = (_Float16*)alloc((size_t)T * QLD * 2);
  _Float16* khp  = (_Float16*)alloc((size_t)TP * KLD * 2);
  _Float16* vth  = (_Float16*)alloc((size_t)KLD * TP * 2);
  _Float16* ctxh = (_Float16*)alloc((size_t)T * QLD * 2);

  // 1) convert inputs to f16
  auto cvt = [&](const float* s, _Float16* d, int n) {
    cvt_f32_f16<<<n / 1024, 256, 0, stream>>>(s, d, n);
  };
  cvt(x,  xh,  T * DM);
  cvt(wq, wqh, QLD * DM);
  cvt(wk, wkh, KLD * DM);
  cvt(wv, wvh, KLD * DM);
  cvt(wo, woh, DM * QLD);

  // 2) QKV projections (WMMA). V is stored transposed for the attention pass.
  gemm_f16_wmma<<<(T / 32) * (QLD / 64) / 8, 256, 0, stream>>>(
      xh, wqh, qh, T, QLD, DM, QLD, /*mode f16*/ 1);
  gemm_f16_wmma<<<(T / 32) * (KLD / 64) / 8, 256, 0, stream>>>(
      xh, wkh, khp, T, KLD, DM, KLD, /*mode f16*/ 1);
  gemm_f16_wmma<<<(T / 32) * (KLD / 64) / 8, 256, 0, stream>>>(
      xh, wvh, vth, T, KLD, DM, TP, /*mode f16 transposed*/ 2);
  vpad_zero<<<(KLD * 32) / 256, 256, 0, stream>>>(vth);

  // 3) fused RMSNorm + RoPE (in place); K also zero-fills its 32 pad rows.
  norm_rope_f16<<<(T * HQ) / 4, 128, 0, stream>>>(qh, pos, qnw, HQ, QLD, T, T);
  norm_rope_f16<<<(TP * HK) / 4, 128, 0, stream>>>(khp, pos, knw, HK, KLD, T, TP);

  // 4) flash sliding-window attention
  attn_wmma<<<(T / 16) * HQ / 8, 256, 0, stream>>>(qh, khp, vth, ctxh);

  // 5) output projection -> f32 d_out
  gemm_f16_wmma<<<(T / 32) * (DM / 64) / 8, 256, 0, stream>>>(
      ctxh, woh, d_out, T, DM, QLD, DM, /*mode f32*/ 0);
}
